// ViST_56453050139023
// MI455X (gfx1250) — matrix operand order repaired and checked by reading, not compiled, not run on hardware
//
#include <hip/hip_runtime.h>
#include <cstdint>
#include <cstddef>

#define B_   32
#define E_   512
#define N_   4096
#define OUT_ 224

typedef __attribute__((ext_vector_type(16))) __bf16 v16bf;
typedef __attribute__((ext_vector_type(8)))  float  v8f;

union FragU { uint4 q[2]; v16bf v; };

// ---------- K1: per-(b,e) mean over N and 1/(||x-mean||+1e-8) ----------
__global__ void row_stats_kernel(const float* __restrict__ x,
                                 float* __restrict__ mean,
                                 float* __restrict__ invn) {
  __shared__ float s_sum[256];
  __shared__ float s_sq[256];
  int row = blockIdx.x;                        // b*E + e  (0..16383)
  const float4* rp = (const float4*)(x + (size_t)row * N_);
  int t = threadIdx.x;
  float s = 0.f, q = 0.f;
#pragma unroll
  for (int i = 0; i < 4; ++i) {
    float4 v = rp[t + i * 256];
    s += v.x + v.y + v.z + v.w;
    q += v.x * v.x + v.y * v.y + v.z * v.z + v.w * v.w;
  }
  s_sum[t] = s; s_sq[t] = q;
  __syncthreads();
  for (int off = 128; off > 0; off >>= 1) {
    if (t < off) { s_sum[t] += s_sum[t + off]; s_sq[t] += s_sq[t + off]; }
    __syncthreads();
  }
  if (t == 0) {
    float mu = s_sum[0] * (1.0f / (float)N_);
    float ss = s_sq[0] - (float)N_ * mu * mu;
    ss = ss > 0.f ? ss : 0.f;
    mean[row] = mu;
    invn[row] = 1.0f / (sqrtf(ss) + 1e-8f);
  }
}

// ---------- K2: per-(b,n) mean over E ----------
__global__ void col_mean_kernel(const float* __restrict__ x,
                                float* __restrict__ smean) {
  int bid = blockIdx.x;                        // 32 * 16
  int b = bid >> 4;
  int n = ((bid & 15) << 8) + threadIdx.x;
  const float* p = x + (size_t)b * E_ * N_ + n;
  float s = 0.f;
  for (int e = 0; e < E_; ++e) s += p[(size_t)e * N_];
  smean[b * N_ + n] = s * (1.0f / (float)E_);
}

// ---------- K3: xn = (x - mean) * invnorm -> bf16 ----------
__device__ __forceinline__ unsigned f2bf(float f) {
  unsigned u = __float_as_uint(f);
  unsigned r = u + 0x7FFFu + ((u >> 16) & 1u);   // round-to-nearest-even
  return r >> 16;
}

__global__ void normalize_bf16_kernel(const float* __restrict__ x,
                                      const float* __restrict__ mean,
                                      const float* __restrict__ invn,
                                      unsigned short* __restrict__ xn) {
  size_t idx4 = (size_t)blockIdx.x * 256 + threadIdx.x;  // float4 index
  int row = (int)(idx4 >> 10);                            // 1024 float4 per row
  float mu = mean[row], iv = invn[row];
  float4 v = ((const float4*)x)[idx4];
  uint2 o;
  o.x = f2bf((v.x - mu) * iv) | (f2bf((v.y - mu) * iv) << 16);
  o.y = f2bf((v.z - mu) * iv) | (f2bf((v.w - mu) * iv) << 16);
  ((uint2*)xn)[idx4] = o;
}

// ---------- K4: corr[b] = Xn[b] * Xn[b]^T via WMMA bf16 ----------
// A 16x32 frag (lane = M%16, h = lane/16): b128 chunks at k = kb+8h and kb+16+8h
__device__ __forceinline__ FragU load_fragA(const unsigned short* row, int kb, int h) {
  FragU f;
  f.q[0] = *(const uint4*)(row + kb + h * 8);
  f.q[1] = *(const uint4*)(row + kb + 16 + h * 8);
  return f;
}
// B 32x16 frag (lane = N%16, h = lane/16): b128 chunks at k = kb+8h and kb+16+8h
__device__ __forceinline__ FragU load_fragB(const unsigned short* row, int kb, int h) {
  FragU f;
  f.q[0] = *(const uint4*)(row + kb + h * 8);
  f.q[1] = *(const uint4*)(row + kb + 16 + h * 8);
  return f;
}

// Each wave computes a 64x32 tile of corr (4x2 WMMA subtiles): 12 b128 loads
// per 8 WMMAs (1.5 loads/WMMA). ~140 VGPRs total -> no spills.
// Block = 8 waves (2 row-waves x 4 col-waves) -> 128x128 block tile.
__global__ void corr_wmma_kernel(const unsigned short* __restrict__ xn,
                                 float* __restrict__ corr) {
  int lane = threadIdx.x & 31;
  int wave = threadIdx.x >> 5;               // 0..7
  int wy = wave & 1;                         // 2 row-waves of 64
  int wx = wave >> 1;                        // 4 col-waves of 32
  int b  = blockIdx.z;
  int i0 = blockIdx.y * 128 + wy * 64;
  int j0 = blockIdx.x * 128 + wx * 32;
  int h  = lane >> 4;
  int l  = lane & 15;

  const unsigned short* base = xn + (size_t)b * E_ * N_;
  const unsigned short* ra[4];
  const unsigned short* rb[2];
#pragma unroll
  for (int t = 0; t < 4; ++t) ra[t] = base + (size_t)(i0 + 16 * t + l) * N_;
#pragma unroll
  for (int t = 0; t < 2; ++t) rb[t] = base + (size_t)(j0 + 16 * t + l) * N_;

  v8f acc[4][2];
#pragma unroll
  for (int ti = 0; ti < 4; ++ti)
#pragma unroll
    for (int tj = 0; tj < 2; ++tj)
      acc[ti][tj] = (v8f){};

  for (int kb = 0; kb < N_; kb += 32) {
    FragU af[4], bf[2];
#pragma unroll
    for (int t = 0; t < 4; ++t) af[t] = load_fragA(ra[t], kb, h);
#pragma unroll
    for (int t = 0; t < 2; ++t) bf[t] = load_fragB(rb[t], kb, h);
#pragma unroll
    for (int ti = 0; ti < 4; ++ti)
#pragma unroll
      for (int tj = 0; tj < 2; ++tj)
        acc[ti][tj] = __builtin_amdgcn_wmma_f32_16x16x32_bf16(
            false, af[ti].v, false, bf[tj].v, (short)0, acc[ti][tj], false, false);
  }

  // C/D layout: lanes 0-15 VGPR r -> M=r; lanes 16-31 -> M=r+8; N = lane&15
  float* crow = corr + (size_t)b * E_ * E_;
#pragma unroll
  for (int ti = 0; ti < 4; ++ti)
#pragma unroll
    for (int tj = 0; tj < 2; ++tj) {
      float* dst = crow + (size_t)(i0 + 16 * ti) * E_ + (j0 + 16 * tj);
#pragma unroll
      for (int r = 0; r < 8; ++r)
        dst[(size_t)(r + h * 8) * E_ + l] = acc[ti][tj][r];
    }
}

// ---------- min/max helpers: order-preserving uint encoding ----------
__device__ __forceinline__ unsigned fenc(float f) {
  unsigned u = __float_as_uint(f);
  return (u & 0x80000000u) ? ~u : (u | 0x80000000u);
}
__device__ __forceinline__ float fdec(unsigned u) {
  unsigned b = (u & 0x80000000u) ? (u & 0x7FFFFFFFu) : ~u;
  return __uint_as_float(b);
}

__global__ void init_mm_kernel(unsigned* mm) {
  int t = threadIdx.x;
  if (t < 3) { mm[t * 2] = 0xFFFFFFFFu; mm[t * 2 + 1] = 0u; }
}

// ---------- K6-8: bilinear resize (align_corners=False) + block min/max ----------
// src value(gy,gx) = (g = gy*gs+gx) < nvalid ? src[b*nvalid + g] : 0
__global__ void resize_kernel(const float* __restrict__ src,
                              float* __restrict__ out,
                              unsigned* __restrict__ mm,
                              int gs, int nvalid, int ch) {
  __shared__ unsigned s_mn[256], s_mx[256];
  int idx = blockIdx.x * 256 + threadIdx.x;     // < B*224*224 (exact)
  int xc = idx % OUT_;
  int yc = (idx / OUT_) % OUT_;
  int b  = idx / (OUT_ * OUT_);
  float scale = (float)gs / (float)OUT_;
  float ry = fmaxf(((float)yc + 0.5f) * scale - 0.5f, 0.f);
  float rx = fmaxf(((float)xc + 0.5f) * scale - 0.5f, 0.f);
  int y0 = (int)floorf(ry); y0 = y0 < gs - 1 ? y0 : gs - 1;
  int x0 = (int)floorf(rx); x0 = x0 < gs - 1 ? x0 : gs - 1;
  int y1 = y0 + 1 < gs - 1 ? y0 + 1 : gs - 1;
  int x1 = x0 + 1 < gs - 1 ? x0 + 1 : gs - 1;
  float wy = ry - (float)y0;
  float wx = rx - (float)x0;
  const float* sb = src + (size_t)b * nvalid;
  int g00 = y0 * gs + x0, g01 = y0 * gs + x1, g10 = y1 * gs + x0, g11 = y1 * gs + x1;
  float f00 = g00 < nvalid ? sb[g00] : 0.f;
  float f01 = g01 < nvalid ? sb[g01] : 0.f;
  float f10 = g10 < nvalid ? sb[g10] : 0.f;
  float f11 = g11 < nvalid ? sb[g11] : 0.f;
  float v = (f00 * (1.f - wx) + f01 * wx) * (1.f - wy)
          + (f10 * (1.f - wx) + f11 * wx) * wy;
  out[((size_t)(b * 3 + ch) * OUT_ + yc) * OUT_ + xc] = v;

  unsigned e = fenc(v);
  s_mn[threadIdx.x] = e; s_mx[threadIdx.x] = e;
  __syncthreads();
  for (int off = 128; off > 0; off >>= 1) {
    if (threadIdx.x < off) {
      unsigned a = s_mn[threadIdx.x + off];
      unsigned m = s_mx[threadIdx.x + off];
      if (a < s_mn[threadIdx.x]) s_mn[threadIdx.x] = a;
      if (m > s_mx[threadIdx.x]) s_mx[threadIdx.x] = m;
    }
    __syncthreads();
  }
  if (threadIdx.x == 0) {
    atomicMin(&mm[ch * 2],     s_mn[0]);
    atomicMax(&mm[ch * 2 + 1], s_mx[0]);
  }
}

// ---------- K9: in-place global min-max normalize ----------
__global__ void final_norm_kernel(float* __restrict__ out,
                                  const unsigned* __restrict__ mm) {
  int idx = blockIdx.x * 256 + threadIdx.x;   // < B*3*224*224 (exact)
  int ch = (idx / (OUT_ * OUT_)) % 3;
  float mn = fdec(mm[ch * 2]);
  float mx = fdec(mm[ch * 2 + 1]);
  float d = mx - mn;
  float v = out[idx];
  out[idx] = (d < 1e-8f) ? 0.0f : (v - mn) / (d + 1e-8f);
}

extern "C" void kernel_launch(void* const* d_in, const int* in_sizes, int n_in,
                              void* d_out, int out_size, void* d_ws, size_t ws_size,
                              hipStream_t stream) {
  const float* x = (const float*)d_in[0];      // [32, 512, 4096, 1] fp32
  float* out = (float*)d_out;                  // [32, 3, 224, 224] fp32

  // Workspace layout (~161 MB total)
  char* ws = (char*)d_ws;
  float* mean  = (float*)ws;                   // 32*512 fp32
  float* invn  = mean + B_ * E_;               // 32*512 fp32
  float* smean = invn + B_ * E_;               // 32*4096 fp32
  unsigned* mm = (unsigned*)(smean + B_ * N_); // 6 uints
  float* corr = (float*)(ws + (size_t)(1u << 20));                      // 32 MB
  unsigned short* xn = (unsigned short*)(ws + (size_t)(1u << 20)
                        + (size_t)B_ * E_ * E_ * sizeof(float));        // 128 MB

  // K1: row stats
  row_stats_kernel<<<B_ * E_, 256, 0, stream>>>(x, mean, invn);
  // K2: column means
  col_mean_kernel<<<B_ * (N_ / 256), 256, 0, stream>>>(x, smean);
  // K3: normalize + bf16 quantize
  normalize_bf16_kernel<<<(int)(((size_t)B_ * E_ * N_ / 4) / 256), 256, 0, stream>>>(
      x, mean, invn, xn);
  // K4: batched correlation GEMM (WMMA bf16 -> f32), 64x32 per wave
  dim3 ggrid(E_ / 128, E_ / 128, B_);
  corr_wmma_kernel<<<ggrid, 256, 0, stream>>>(xn, corr);
  // K5: init min/max slots
  init_mm_kernel<<<1, 32, 0, stream>>>(mm);
  // K6-8: resize three views to 224x224, tracking global min/max per view
  int rgrid = (B_ * OUT_ * OUT_) / 256;
  resize_kernel<<<rgrid, 256, 0, stream>>>(mean,  out, mm, 23,  E_,      0); // temporal
  resize_kernel<<<rgrid, 256, 0, stream>>>(smean, out, mm, 64,  N_,      1); // spatial
  resize_kernel<<<rgrid, 256, 0, stream>>>(corr,  out, mm, E_,  E_ * E_, 2); // correlation
  // K9: global min-max normalize in place
  final_norm_kernel<<<(B_ * 3 * OUT_ * OUT_) / 256, 256, 0, stream>>>(out, mm);
}
